// SelfConditionedMultiheadAttention_2542620639588
// MI455X (gfx1250) — matrix operand order repaired and checked by reading, not compiled, not run on hardware
//
#include <hip/hip_runtime.h>

// ---------------------------------------------------------------------------
// SelfConditionedMultiheadAttention for MI455X (gfx1250, wave32, WMMA)
// B=2, T=2048, D=1024, H=16, Hd=64
// ---------------------------------------------------------------------------

typedef _Float16 h8  __attribute__((ext_vector_type(8)));
typedef _Float16 h16 __attribute__((ext_vector_type(16)));
typedef float    f8  __attribute__((ext_vector_type(8)));

static constexpr int BATCH = 2;
static constexpr int TSEQ  = 2048;
static constexpr int DIM   = 1024;
static constexpr int NHEAD = 16;
static constexpr int HDIM  = 64;
static constexpr int NROWS = BATCH * TSEQ; // 4096

__device__ __forceinline__ f8 wmma_f16(h16 a, h16 b, f8 c) {
    return __builtin_amdgcn_wmma_f32_16x16x32_f16(
        false, a, false, b, (short)0, c, false, false);
}

// A fragment (16x32, row-major): a[e] = A[m_tile+m][k0 + half*16 + e]
__device__ __forceinline__ h16 load_a_frag(const _Float16* A, int lda,
                                           int m_tile, int k0, int lane) {
    int m = lane & 15, half = lane >> 4;
    const _Float16* p = A + (size_t)(m_tile + m) * lda + k0 + half * 16;
    h8 lo = *(const h8*)(p);
    h8 hi = *(const h8*)(p + 8);
    h16 a;
#pragma unroll
    for (int i = 0; i < 8; ++i) { a[i] = lo[i]; a[i + 8] = hi[i]; }
    return a;
}

// B fragment (32x16) from Bt[n][k] row-major (= B col-major), ldb = K:
// b[e] = Bt[n_tile+n][k0 + half*16 + e]  -> contiguous 32 bytes per lane
__device__ __forceinline__ h16 load_b_frag(const _Float16* Bt, int ldb,
                                           int n_tile, int k0, int lane) {
    int n = lane & 15, half = lane >> 4;
    const _Float16* p = Bt + (size_t)(n_tile + n) * ldb + k0 + half * 16;
    h8 lo = *(const h8*)(p);
    h8 hi = *(const h8*)(p + 8);
    h16 b;
#pragma unroll
    for (int i = 0; i < 8; ++i) { b[i] = lo[i]; b[i + 8] = hi[i]; }
    return b;
}

__device__ __forceinline__ h16 pack_h16(h8 lo, h8 hi) {
    h16 r;
#pragma unroll
    for (int i = 0; i < 8; ++i) { r[i] = lo[i]; r[i + 8] = hi[i]; }
    return r;
}

// ---------------------------------------------------------------------------
// Elementwise conversion kernels
// ---------------------------------------------------------------------------
__global__ void cvt_f16_kernel(const float* __restrict__ x,
                               _Float16* __restrict__ y, int n) {
    int i = blockIdx.x * blockDim.x + threadIdx.x;
    if (i < n) y[i] = (_Float16)x[i];
}

// Wt[n*D + k] = W[k*D + n]  (f32 row-major -> f16 column-major)
__global__ void cvt_transpose_f16_kernel(const float* __restrict__ W,
                                         _Float16* __restrict__ Wt, int D) {
    int o = blockIdx.x * blockDim.x + threadIdx.x;
    if (o < D * D) {
        int n = o / D, k = o - n * D;
        Wt[o] = (_Float16)W[(size_t)k * D + n];
    }
}

// b_qs[n] = sum_k b_q[k] * W_s[k][n]
__global__ void bias_qs_kernel(const float* __restrict__ bq,
                               const float* __restrict__ Ws,
                               float* __restrict__ out, int D) {
    int n = blockIdx.x * blockDim.x + threadIdx.x;
    if (n < D) {
        float s = 0.f;
        for (int k = 0; k < D; ++k) s += bq[k] * Ws[(size_t)k * D + n];
        out[n] = s;
    }
}

// ---------------------------------------------------------------------------
// WMMA GEMM, software-pipelined (double-buffered fragments, K-step 64).
//   MODE 0: C f16 row-major; MODE 1: C f16 transposed; MODE 2: C f32 + bias
// Block 128 thr = 4 waves; wave tile 64x32; block tile 64x128. Grid (M/64,N/128)
// ---------------------------------------------------------------------------
template <int MODE>
__global__ __launch_bounds__(128) void gemm_wmma_kernel(
    const _Float16* __restrict__ A, const _Float16* __restrict__ Bt,
    void* __restrict__ Cout, const float* __restrict__ bias,
    int M, int N, int K) {
    const int lane = threadIdx.x & 31;
    const int wave = threadIdx.x >> 5;
    const int m0 = blockIdx.x * 64;
    const int n0 = blockIdx.y * 128 + wave * 32;

    f8 acc[4][2] = {};
    h16 a0[4], b0[2], a1[4], b1[2];

#pragma unroll
    for (int mi = 0; mi < 4; ++mi) a0[mi] = load_a_frag(A, K, m0 + mi * 16, 0, lane);
#pragma unroll
    for (int ni = 0; ni < 2; ++ni) b0[ni] = load_b_frag(Bt, K, n0 + ni * 16, 0, lane);

    for (int k0 = 0; k0 < K; k0 += 64) {
        // prefetch k0+32 while computing k0
#pragma unroll
        for (int mi = 0; mi < 4; ++mi) a1[mi] = load_a_frag(A, K, m0 + mi * 16, k0 + 32, lane);
#pragma unroll
        for (int ni = 0; ni < 2; ++ni) b1[ni] = load_b_frag(Bt, K, n0 + ni * 16, k0 + 32, lane);
#pragma unroll
        for (int mi = 0; mi < 4; ++mi)
#pragma unroll
            for (int ni = 0; ni < 2; ++ni)
                acc[mi][ni] = wmma_f16(a0[mi], b0[ni], acc[mi][ni]);

        if (k0 + 64 < K) { // prefetch k0+64 while computing k0+32
#pragma unroll
            for (int mi = 0; mi < 4; ++mi) a0[mi] = load_a_frag(A, K, m0 + mi * 16, k0 + 64, lane);
#pragma unroll
            for (int ni = 0; ni < 2; ++ni) b0[ni] = load_b_frag(Bt, K, n0 + ni * 16, k0 + 64, lane);
        }
#pragma unroll
        for (int mi = 0; mi < 4; ++mi)
#pragma unroll
            for (int ni = 0; ni < 2; ++ni)
                acc[mi][ni] = wmma_f16(a1[mi], b1[ni], acc[mi][ni]);
    }

    const int cn = lane & 15, half = lane >> 4;
#pragma unroll
    for (int mi = 0; mi < 4; ++mi) {
#pragma unroll
        for (int ni = 0; ni < 2; ++ni) {
            const int col = n0 + ni * 16 + cn;
            const float bv = bias ? bias[col] : 0.f;
#pragma unroll
            for (int r = 0; r < 8; ++r) {
                const int row = m0 + mi * 16 + half * 8 + r;
                const float v = acc[mi][ni][r] + bv;
                if (MODE == 0)
                    ((_Float16*)Cout)[(size_t)row * N + col] = (_Float16)v;
                else if (MODE == 1)
                    ((_Float16*)Cout)[(size_t)col * M + row] = (_Float16)v;
                else
                    ((float*)Cout)[(size_t)row * N + col] = v;
            }
        }
    }
}

// ---------------------------------------------------------------------------
// Flash attention. 4 waves x 16 q-rows = 64 q-rows per block; key chunk = 64.
// K chunk is DMA-staged into LDS one chunk ahead with
// GLOBAL_LOAD_ASYNC_TO_LDS_B128 (ASYNCcnt), double-buffered.
// Row-sums of P come from a WMMA against an all-ones B matrix (replicated
// across lanes in C layout == exactly the online-softmax broadcast we need).
// Grid: (T/64, H, B), block 128.
// ---------------------------------------------------------------------------
__global__ __launch_bounds__(128) void attn_kernel(
    const _Float16* __restrict__ Q, const _Float16* __restrict__ Kf,
    const _Float16* __restrict__ V, _Float16* __restrict__ O) {
    __shared__ __attribute__((aligned(16))) _Float16 Krm[2][64][72]; // K chunk, row-major, dbl-buffered
    __shared__ __attribute__((aligned(16))) _Float16 Vt[64][72];     // V chunk transposed [hd][kk]
    __shared__ __attribute__((aligned(16))) _Float16 Pb[4][16][72];  // per-wave P tile [m][k]

    const int lane = threadIdx.x & 31;
    const int wave = threadIdx.x >> 5;
    const int m = lane & 15, half = lane >> 4;
    const int b = blockIdx.z, h = blockIdx.y;
    const int q0 = blockIdx.x * 64 + wave * 16;

    const size_t base = (size_t)b * TSEQ * DIM + (size_t)h * HDIM;
    const _Float16* Qh = Q + base;
    const _Float16* Kh = Kf + base;
    const _Float16* Vh = V + base;
    _Float16* Oh = O + base;
    const unsigned long long kbase = (unsigned long long)(size_t)Kh;

    // async DMA of one 64x64 f16 K chunk into LDS buffer `buf` (512 x 16B tasks)
    auto issueK = [&](int kc, int buf) {
        for (int t = threadIdx.x; t < 512; t += 128) {
            const int row = t >> 3, g = t & 7;
            unsigned lds = (unsigned)(size_t)&Krm[buf][row][g * 8];
            unsigned voff = (unsigned)(((kc + row) * DIM + g * 8) * 2);
            asm volatile("global_load_async_to_lds_b128 %0, %1, %2"
                         :: "v"(lds), "v"(voff), "s"(kbase) : "memory");
        }
    };

    h16 qa0 = load_a_frag(Qh, DIM, q0, 0, lane);
    h16 qa1 = load_a_frag(Qh, DIM, q0, 32, lane);

    h16 ones;
#pragma unroll
    for (int i = 0; i < 16; ++i) ones[i] = (_Float16)1.0f;

    f8 o[4] = {};
    float mrow[8], lrow[8];
#pragma unroll
    for (int r = 0; r < 8; ++r) { mrow[r] = -1e30f; lrow[r] = 0.f; }

    const float sc = 0.125f; // 1/sqrt(64)

    issueK(0, 0);

    for (int ci = 0; ci < TSEQ / 64; ++ci) {
        const int kc = ci * 64;
        const int cur = ci & 1;

        asm volatile("s_wait_asynccnt 0" ::: "memory");
        __syncthreads(); // K chunk `cur` resident block-wide; prior-iter LDS reads done
        if (kc + 64 < TSEQ) issueK(kc + 64, cur ^ 1); // overlap DMA with compute

        // ---- stage V chunk transposed into LDS (512 x 16B tasks) ----
        for (int t = threadIdx.x; t < 512; t += 128) {
            const int kk = t >> 3, c8 = (t & 7) * 8;
            h8 v = *(const h8*)(Vh + (size_t)(kc + kk) * DIM + c8);
#pragma unroll
            for (int i = 0; i < 8; ++i) Vt[c8 + i][kk] = v[i];
        }
        __syncthreads();

        if (kc + 64 < TSEQ)
            __builtin_prefetch(Vh + (size_t)(kc + 64) * DIM, 0, 1);

        // ---- S = Q @ K^T for 64 keys (4 C-frags), K-frags from LDS ----
        f8 s[4] = {};
#pragma unroll
        for (int kt = 0; kt < 4; ++kt) {
            h16 kb0 = pack_h16(*(const h8*)&Krm[cur][kt * 16 + m][half * 16],
                               *(const h8*)&Krm[cur][kt * 16 + m][half * 16 + 8]);
            h16 kb1 = pack_h16(*(const h8*)&Krm[cur][kt * 16 + m][32 + half * 16],
                               *(const h8*)&Krm[cur][kt * 16 + m][32 + half * 16 + 8]);
            s[kt] = wmma_f16(qa0, kb0, s[kt]);
            s[kt] = wmma_f16(qa1, kb1, s[kt]);
        }

        // ---- online softmax: running max (shuffle butterfly) + exp ----
        float alpha[8];
#pragma unroll
        for (int r = 0; r < 8; ++r) {
            float v0 = s[0][r] * sc, v1 = s[1][r] * sc;
            float v2 = s[2][r] * sc, v3 = s[3][r] * sc;
            float mx = fmaxf(fmaxf(v0, v1), fmaxf(v2, v3));
#pragma unroll
            for (int off = 1; off < 16; off <<= 1)
                mx = fmaxf(mx, __shfl_xor(mx, off, 32));
            const float mn = fmaxf(mrow[r], mx);
            alpha[r] = __expf(mrow[r] - mn);
            mrow[r] = mn;
            s[0][r] = __expf(v0 - mn);
            s[1][r] = __expf(v1 - mn);
            s[2][r] = __expf(v2 - mn);
            s[3][r] = __expf(v3 - mn);
        }

        // ---- P (C layout) -> LDS -> A layout ----
#pragma unroll
        for (int r = 0; r < 8; ++r) {
            const int row = half * 8 + r;
#pragma unroll
            for (int kt = 0; kt < 4; ++kt)
                Pb[wave][row][kt * 16 + m] = (_Float16)s[kt][r];
        }
        asm volatile("s_wait_dscnt 0" ::: "memory"); // same-wave DS RAW fence
        h16 pf0 = pack_h16(*(const h8*)&Pb[wave][m][half * 16],
                           *(const h8*)&Pb[wave][m][half * 16 + 8]);
        h16 pf1 = pack_h16(*(const h8*)&Pb[wave][m][32 + half * 16],
                           *(const h8*)&Pb[wave][m][32 + half * 16 + 8]);

        // ---- row sums via WMMA with all-ones B (replicated across lanes) ----
        f8 sums = {};
        sums = wmma_f16(pf0, ones, sums);
        sums = wmma_f16(pf1, ones, sums);
#pragma unroll
        for (int r = 0; r < 8; ++r) lrow[r] = lrow[r] * alpha[r] + sums[r];

        // ---- rescale accumulators, then O += P @ V_chunk ----
#pragma unroll
        for (int j = 0; j < 4; ++j)
#pragma unroll
            for (int r = 0; r < 8; ++r) o[j][r] *= alpha[r];

#pragma unroll
        for (int j = 0; j < 4; ++j) {
            h16 vb0 = pack_h16(*(const h8*)&Vt[j * 16 + m][half * 16],
                               *(const h8*)&Vt[j * 16 + m][half * 16 + 8]);
            h16 vb1 = pack_h16(*(const h8*)&Vt[j * 16 + m][32 + half * 16],
                               *(const h8*)&Vt[j * 16 + m][32 + half * 16 + 8]);
            o[j] = wmma_f16(pf0, vb0, o[j]);
            o[j] = wmma_f16(pf1, vb1, o[j]);
        }
    }

    // ---- epilogue: normalize and store f16 head-slice ----
#pragma unroll
    for (int r = 0; r < 8; ++r) {
        const float inv = 1.f / lrow[r];
        const int row = q0 + half * 8 + r;
#pragma unroll
        for (int j = 0; j < 4; ++j)
            Oh[(size_t)row * DIM + j * 16 + m] = (_Float16)(o[j][r] * inv);
    }
}

// ---------------------------------------------------------------------------
// Host-side launch
// ---------------------------------------------------------------------------
extern "C" void kernel_launch(void* const* d_in, const int* in_sizes, int n_in,
                              void* d_out, int out_size, void* d_ws, size_t ws_size,
                              hipStream_t stream) {
    const float* x   = (const float*)d_in[0];
    const float* W_q = (const float*)d_in[1];
    const float* b_q = (const float*)d_in[2];
    const float* W_k = (const float*)d_in[3];
    const float* b_k = (const float*)d_in[4];
    const float* W_v = (const float*)d_in[5];
    const float* b_v = (const float*)d_in[6];
    const float* W_o = (const float*)d_in[7];
    const float* b_o = (const float*)d_in[8];
    const float* W_s = (const float*)d_in[9];

    const size_t MB = 1024u * 1024u;
    char* ws = (char*)d_ws;
    _Float16* Xh   = (_Float16*)(ws + 0 * MB);   // 4096x1024 f16 (8MB)
    _Float16* Wqh  = (_Float16*)(ws + 8 * MB);   // W_q row-major f16
    _Float16* Wst  = (_Float16*)(ws + 10 * MB);  // W_s^T f16
    _Float16* Wkt  = (_Float16*)(ws + 12 * MB);  // W_k^T f16
    _Float16* Wvt  = (_Float16*)(ws + 14 * MB);  // W_v^T f16
    _Float16* Wot  = (_Float16*)(ws + 16 * MB);  // W_o^T f16
    _Float16* Wqst = (_Float16*)(ws + 18 * MB);  // (W_q@W_s)^T f16
    _Float16* Qh   = (_Float16*)(ws + 20 * MB);  // 8MB
    _Float16* Kh   = (_Float16*)(ws + 28 * MB);  // 8MB
    _Float16* Vh   = (_Float16*)(ws + 36 * MB);  // 8MB
    _Float16* Oh   = (_Float16*)(ws + 44 * MB);  // 8MB
    float*    bqs  = (float*)   (ws + 52 * MB);  // 4KB

    const int D = DIM, NR = NROWS;

    cvt_f16_kernel<<<(NR * D + 255) / 256, 256, 0, stream>>>(x, Xh, NR * D);
    cvt_f16_kernel<<<(D * D + 255) / 256, 256, 0, stream>>>(W_q, Wqh, D * D);
    cvt_transpose_f16_kernel<<<(D * D + 255) / 256, 256, 0, stream>>>(W_s, Wst, D);
    cvt_transpose_f16_kernel<<<(D * D + 255) / 256, 256, 0, stream>>>(W_k, Wkt, D);
    cvt_transpose_f16_kernel<<<(D * D + 255) / 256, 256, 0, stream>>>(W_v, Wvt, D);
    cvt_transpose_f16_kernel<<<(D * D + 255) / 256, 256, 0, stream>>>(W_o, Wot, D);
    bias_qs_kernel<<<(D + 255) / 256, 256, 0, stream>>>(b_q, W_s, bqs, D);

    // fold conditioning: W_qs = W_q @ W_s, stored transposed
    gemm_wmma_kernel<1><<<dim3(D / 64, D / 128), 128, 0, stream>>>(
        Wqh, Wst, (void*)Wqst, nullptr, D, D, D);

    // projections
    gemm_wmma_kernel<0><<<dim3(NR / 64, D / 128), 128, 0, stream>>>(
        Xh, Wqst, (void*)Qh, bqs, NR, D, D);
    gemm_wmma_kernel<0><<<dim3(NR / 64, D / 128), 128, 0, stream>>>(
        Xh, Wkt, (void*)Kh, b_k, NR, D, D);
    gemm_wmma_kernel<0><<<dim3(NR / 64, D / 128), 128, 0, stream>>>(
        Xh, Wvt, (void*)Vh, b_v, NR, D, D);

    // flash attention
    attn_kernel<<<dim3(TSEQ / 64, NHEAD, BATCH), 128, 0, stream>>>(Qh, Kh, Vh, Oh);

    // output projection -> f32
    gemm_wmma_kernel<2><<<dim3(NR / 64, D / 128), 128, 0, stream>>>(
        Oh, Wot, d_out, b_o, NR, D, D);
}
